// MessagePassingModel_30459908063664
// MI455X (gfx1250) — matrix operand order repaired and checked by reading, not compiled, not run on hardware
//
#include <hip/hip_runtime.h>
#include <hip/hip_bf16.h>
#include <math.h>

#define FDIM   32
#define NNODE  32
#define NBATCH 32
#define BN     (NBATCH*NNODE)          // 1024 nodes
#define EPB    (NNODE*(NNODE-1))       // 992 edges per batch
#define NEDGE  (NBATCH*EPB)            // 31744 edges
#define NROWS  18                      // 2 parities * 9 m-components
#define XSZ    (BN*NROWS*FDIM)         // 589824 floats
#define MROWS  (BN*NROWS)              // 18432 GEMM rows

typedef __attribute__((ext_vector_type(16))) _Float16 v16h;
typedef __attribute__((ext_vector_type(8)))  float    v8f;

__constant__ int c_DM[9] = {0,1,1,1,2,2,2,2,2};

// ---------------------------------------------------------------------------
// Clebsch-Gordan table (matches _build_cg, computed in fp64 on device)
// ---------------------------------------------------------------------------
__device__ double dfact(int n){ double r=1.0; for(int i=2;i<=n;++i) r*=(double)i; return r; }

__device__ double cgc_d(int j1,int m1,int j2,int m2,int J,int M){
  if (m1+m2 != M) return 0.0;
  int ad = j1-j2; if (ad<0) ad=-ad;
  if (J < ad || J > j1+j2) return 0.0;
  double pref = (double)(2*J+1)*dfact(J+j1-j2)*dfact(J-j1+j2)*dfact(j1+j2-J)/dfact(j1+j2+J+1);
  pref *= dfact(J+M)*dfact(J-M)*dfact(j1-m1)*dfact(j1+m1)*dfact(j2-m2)*dfact(j2+m2);
  double s=0.0;
  for (int k=0;k<=j1+j2;++k){
    int d0=j1+j2-J-k, d1=j1-m1-k, d2=j2+m2-k, d3=J-j2+m1+k, d4=J-j1-m2+k;
    if (d0<0||d1<0||d2<0||d3<0||d4<0) continue;
    double denom = dfact(k)*dfact(d0)*dfact(d1)*dfact(d2)*dfact(d3)*dfact(d4);
    s += ((k&1)? -1.0 : 1.0)/denom;
  }
  return sqrt(pref)*s;
}

__device__ void u_entry(int l,int a,int b,double* re,double* im){
  *re=0.0; *im=0.0;
  const double is2 = 0.70710678118654752440;
  if (a==l && b==l){ *re=1.0; return; }
  for (int m=1;m<=l;++m){
    double sgn = (m&1)? -1.0 : 1.0;
    if (a==l+m && b==l+m){ *re=sgn*is2; return; }
    if (a==l+m && b==l-m){ *re=is2;     return; }
    if (a==l-m && b==l-m){ *im=is2;     return; }
    if (a==l-m && b==l+m){ *im=-sgn*is2;return; }
  }
}

__device__ int l_of(int a){ return (a==0)?0:((a<4)?1:2); }

__global__ void k_cg(float* CG){
  int t = threadIdx.x;
  if (t >= 729) return;
  int c = t%9, b = (t/9)%9, a = t/81;
  int l1=l_of(a), l2=l_of(b), l3=l_of(c);
  float out = 0.f;
  int d = l1-l2; if (d<0) d=-d;
  if (!(l3 < d || l3 > l1+l2)){
    int al=a-l1*l1, bl=b-l2*l2, cl=c-l3*l3;
    double gre=0.0, gim=0.0;
    for (int m=0;m<=2*l1;++m){
      double u1r,u1i; u_entry(l1,al,m,&u1r,&u1i);
      if (u1r==0.0 && u1i==0.0) continue;
      for (int n=0;n<=2*l2;++n){
        double u2r,u2i; u_entry(l2,bl,n,&u2r,&u2i);
        if (u2r==0.0 && u2i==0.0) continue;
        double pr = u1r*u2r - u1i*u2i;
        double pi = u1r*u2i + u1i*u2r;
        for (int o=0;o<=2*l3;++o){
          double u3r,u3i; u_entry(l3,cl,o,&u3r,&u3i);
          if (u3r==0.0 && u3i==0.0) continue;
          double cv = cgc_d(l1,m-l1,l2,n-l2,l3,o-l3);
          if (cv==0.0) continue;
          gre += (pr*u3r + pi*u3i)*cv;   // * conj(u3)
          gim += (pi*u3r - pr*u3i)*cv;
        }
      }
    }
    double r = (fabs(gim) > fabs(gre)) ? gim : gre;
    out = (float)r;
  }
  CG[t] = out;
}

__global__ void k_cgpack(const float* CG, int* idx, float* val, int* cnt){
  if (threadIdx.x || blockIdx.x) return;
  int n=0;
  for (int a=0;a<9;++a) for (int b=0;b<9;++b) for (int c=0;c<9;++c){
    float v = CG[(a*9+b)*9+c];
    if (fabsf(v) > 1e-12f){ idx[n]=(a<<8)|(b<<4)|c; val[n]=v; ++n; }
  }
  *cnt = n;
}

// ---------------------------------------------------------------------------
// Edge geometry: r, spherical harmonics (9), radial basis (8)
// ---------------------------------------------------------------------------
__global__ void k_edge(const float* __restrict__ pos, const int* __restrict__ dsti,
                       const int* __restrict__ srci,
                       float* __restrict__ Er, float* __restrict__ Esh,
                       float* __restrict__ Erad){
  int e = blockIdx.x*blockDim.x + threadIdx.x;
  if (e >= NEDGE) return;
  int b = e / EPB, k = e % EPB;
  int dst = b*NNODE + dsti[k];
  int src = b*NNODE + srci[k];
  float dx = pos[src*3+0]-pos[dst*3+0];
  float dy = pos[src*3+1]-pos[dst*3+1];
  float dz = pos[src*3+2]-pos[dst*3+2];
  float r = sqrtf(dx*dx+dy*dy+dz*dz);
  Er[e] = r;
  float inv = 1.f/(r+1e-10f);
  float x = dx*inv, y = dy*inv, z = dz*inv;
  const float s3 = 1.7320508075688772f;
  float sh[9] = {1.f, y, z, x, s3*x*y, s3*y*z, 0.5f*(3.f*z*z-1.f), s3*x*z, 0.5f*s3*(x*x-y*y)};
  #pragma unroll
  for (int a=0;a<9;++a) Esh[e*9+a]=sh[a];
  float f = 1.f/(1.f+r);
  const float binom[8] = {1.f,7.f,21.f,35.f,35.f,21.f,7.f,1.f};
  float u2 = (r/5.f)*(r/5.f);
  float cl = 1.f-u2; if (cl < 1e-6f) cl = 1e-6f;
  float cut = (r < 5.f) ? expf(1.f - 1.f/cl) : 0.f;
  #pragma unroll
  for (int kk=0; kk<8; ++kk){
    float p1=1.f; for (int q=0;q<kk;++q)   p1 *= f;
    float p2=1.f; for (int q=0;q<7-kk;++q) p2 *= (1.f-f);
    Erad[e*8+kk] = binom[kk]*p1*p2*cut;
  }
}

// ---------------------------------------------------------------------------
// Init x (embedding in scalar channel) and xEF (Ef in l=1 channels)
// ---------------------------------------------------------------------------
__global__ void k_init(const int* __restrict__ Z, const float* __restrict__ embed,
                       const float* __restrict__ Ef,
                       float* __restrict__ X, float* __restrict__ XEF){
  int node = blockIdx.x, f = threadIdx.x;
  float* xp = X + node*NROWS*FDIM;
  float* ep = XEF + node*NROWS*FDIM;
  int b = node / NNODE;
  float emb = embed[Z[node]*FDIM + f];
  #pragma unroll
  for (int p=0;p<2;++p)
    #pragma unroll
    for (int a=0;a<9;++a){
      float xv = (p==0 && a==0) ? emb : 0.f;
      float ev = (a>=1 && a<=3) ? Ef[b*3+(a-1)] : 0.f;
      xp[(p*9+a)*FDIM+f] = xv;
      ep[(p*9+a)*FDIM+f] = ev;
    }
}

// ---------------------------------------------------------------------------
// Message passing: per destination node, gather all 31 in-edges (dense graph)
// ---------------------------------------------------------------------------
__global__ void k_msg(const float* __restrict__ X, const float* __restrict__ Esh,
                      const float* __restrict__ Erad, const float* __restrict__ Wmp,
                      const int* __restrict__ cgidx, const float* __restrict__ cgval,
                      const int* __restrict__ cgcnt,
                      float* __restrict__ Y, int maxdeg){
  __shared__ int   s_idx[768];
  __shared__ float s_val[768];
  int node = blockIdx.x, f = threadIdx.x;
  int nnz = *cgcnt;
  for (int t=f; t<nnz; t+=32){ s_idx[t]=cgidx[t]; s_val[t]=cgval[t]; }
  __syncthreads();
  int b = node / NNODE, i = node % NNODE;
  float acc0[9]={0,0,0,0,0,0,0,0,0}, acc1[9]={0,0,0,0,0,0,0,0,0};
  for (int t=0; t<NNODE-1; ++t){
    int j = (t < i) ? t : t+1;
    int e = b*EPB + i*(NNODE-1) + t;
    float Wr[3];
    #pragma unroll
    for (int l=0;l<3;++l){
      float s=0.f;
      #pragma unroll
      for (int n=0;n<8;++n) s += Erad[e*8+n]*Wmp[(l*8+n)*FDIM+f];
      Wr[l]=s;
    }
    float g[9];
    #pragma unroll
    for (int a=0;a<9;++a) g[a] = Esh[e*9+a]*Wr[c_DM[a]];
    const float* xs = X + (b*NNODE+j)*NROWS*FDIM;
    float xs0[9], xs1[9];
    #pragma unroll
    for (int a=0;a<9;++a){ xs0[a]=xs[a*FDIM+f]; xs1[a]=xs[(9+a)*FDIM+f]; }
    for (int t2=0; t2<nnz; ++t2){
      int id = s_idx[t2];
      int a=(id>>8)&15, bb=(id>>4)&15, c=id&15;
      float w = s_val[t2]*g[bb];
      if ((c_DM[bb]&1)==0){ acc0[c]+=w*xs0[a]; acc1[c]+=w*xs1[a]; }
      else                { acc0[c]+=w*xs1[a]; acc1[c]+=w*xs0[a]; }
    }
  }
  float* yp = Y + node*NROWS*FDIM;
  #pragma unroll
  for (int c=0;c<9;++c){
    float m = (c_DM[c] <= maxdeg) ? 1.f : 0.f;
    yp[c*FDIM+f]     = acc0[c]*m;
    yp[(9+c)*FDIM+f] = acc1[c]*m;
  }
}

// ---------------------------------------------------------------------------
// x = silu(x + y)  (gate = sigmoid of updated scalar channel)
// ---------------------------------------------------------------------------
__global__ void k_add_silu(float* __restrict__ X, const float* __restrict__ Y){
  int node = blockIdx.x, f = threadIdx.x;
  float* xp = X + node*NROWS*FDIM;
  const float* yp = Y + node*NROWS*FDIM;
  float v00 = xp[f] + yp[f];
  float gate = 1.f/(1.f+expf(-v00));
  #pragma unroll
  for (int r=0;r<NROWS;++r){
    float v = xp[r*FDIM+f] + yp[r*FDIM+f];
    xp[r*FDIM+f] = v*gate;
  }
}

__global__ void k_silu(float* __restrict__ X){
  int node = blockIdx.x, f = threadIdx.x;
  float* xp = X + node*NROWS*FDIM;
  float gate = 1.f/(1.f+expf(-xp[f]));
  #pragma unroll
  for (int r=0;r<NROWS;++r) xp[r*FDIM+f] *= gate;
}

// ---------------------------------------------------------------------------
// WMMA dense: Y[r,g] = sum_f X[r,f]*W[f,g] (+ bias at rows r%18==0)
// One wave = one 16-row strip x 32 cols, via v_wmma_f32_16x16x32_f16.
//  - W staged once per block into LDS in WMMA B-fragment layout (f16)
//  - A fragment loaded as 4x b128
//  - branch-free bias (row0+grp*8)%18 precomputed; epilogue transposed
//    through LDS and stored as coalesced b128
// ---------------------------------------------------------------------------
__global__ void k_dense(const float* __restrict__ X, const float* __restrict__ W,
                        const float* __restrict__ bias, float* __restrict__ Yo,
                        int apply_bias){
  __shared__ __align__(32) _Float16 sW[2][32][16]; // [ntile][lane][elem]
  __shared__ float sC[4][16][32];                  // per-wave C tile
  int tid  = threadIdx.x;
  int wave = tid >> 5;
  int lane = tid & 31;
  int m    = lane & 15;
  int grp  = lane >> 4;

  // cooperative W -> LDS (fragment layout): 1024 halfs, 8 per thread
  #pragma unroll
  for (int q=0; q<8; ++q){
    int idx = tid*8 + q;          // 0..1023
    int nt  = idx >> 9;
    int l   = (idx >> 4) & 31;
    int e   = idx & 15;
    int kk  = (e & 7) + 8*(l >> 4) + 16*(e >> 3);    // B layout: k = 8*grp + e (e<8), 16 + 8*grp + (e-8) (e>=8)
    sW[nt][l][e] = (_Float16)W[kk*FDIM + nt*16 + (l & 15)];
  }

  int row0 = (blockIdx.x*4 + wave)*16;
  // A fragment: k runs [8*grp, 8*grp+8) and [16+8*grp, 24+8*grp)
  const float4* x4 = (const float4*)(X + (row0+m)*FDIM);
  float4 q0 = x4[2*grp + 0];
  float4 q1 = x4[2*grp + 1];
  float4 q2 = x4[4 + 2*grp + 0];
  float4 q3 = x4[4 + 2*grp + 1];
  v16h a;
  a[0]=(_Float16)q0.x;  a[1]=(_Float16)q0.y;  a[2]=(_Float16)q0.z;  a[3]=(_Float16)q0.w;
  a[4]=(_Float16)q1.x;  a[5]=(_Float16)q1.y;  a[6]=(_Float16)q1.z;  a[7]=(_Float16)q1.w;
  a[8]=(_Float16)q2.x;  a[9]=(_Float16)q2.y;  a[10]=(_Float16)q2.z; a[11]=(_Float16)q2.w;
  a[12]=(_Float16)q3.x; a[13]=(_Float16)q3.y; a[14]=(_Float16)q3.z; a[15]=(_Float16)q3.w;

  int bm = (row0 + grp*8) % NROWS;   // bias hits where bm+j == 0 or 18
  __syncthreads();

  #pragma unroll
  for (int nt=0; nt<2; ++nt){
    v16h bfr = *(const v16h*)&sW[nt][lane][0];
    float bval = 0.f;
    if (apply_bias) bval = bias[nt*16 + m];
    v8f c = {};
    c = __builtin_amdgcn_wmma_f32_16x16x32_f16(false, a, false, bfr,
                                               (short)0, c, false, false);
    #pragma unroll
    for (int j=0; j<8; ++j){
      int s = bm + j;
      float v = c[j] + ((s==0 || s==NROWS) ? bval : 0.f);
      sC[wave][grp*8+j][nt*16+m] = v;
    }
  }
  __syncthreads();

  // coalesced epilogue: lane -> (row = lane/2, half-row = lane&1), 4x b128
  int r = lane >> 1, half = lane & 1;
  const float4* srcp = (const float4*)&sC[wave][r][half*16];
  float4* dstp = (float4*)&Yo[(row0+r)*FDIM + half*16];
  #pragma unroll
  for (int q=0; q<4; ++q) dstp[q] = srcp[q];
}

// ---------------------------------------------------------------------------
// xEF = couple(x, xEF, Wt, 2);  x += xEF   (fused)
// ---------------------------------------------------------------------------
__global__ void k_couple_ef(float* __restrict__ X, float* __restrict__ XEF,
                            const float* __restrict__ Wt,
                            const int* __restrict__ cgidx, const float* __restrict__ cgval,
                            const int* __restrict__ cgcnt){
  __shared__ int   s_idx[768];
  __shared__ float s_val[768];
  int node = blockIdx.x, f = threadIdx.x;
  int nnz = *cgcnt;
  for (int t=f; t<nnz; t+=32){ s_idx[t]=cgidx[t]; s_val[t]=cgval[t]; }
  __syncthreads();
  float* xp = X + node*NROWS*FDIM;
  float* ep = XEF + node*NROWS*FDIM;
  float x0[9],x1[9],e0[9],e1[9];
  #pragma unroll
  for (int a=0;a<9;++a){
    x0[a]=xp[a*FDIM+f]; x1[a]=xp[(9+a)*FDIM+f];
    e0[a]=ep[a*FDIM+f]; e1[a]=ep[(9+a)*FDIM+f];
  }
  float p0[9]={0,0,0,0,0,0,0,0,0}, p1[9]={0,0,0,0,0,0,0,0,0};
  for (int t=0;t<nnz;++t){
    int id=s_idx[t]; int a=(id>>8)&15, b=(id>>4)&15, c=id&15;
    float w = s_val[t]*Wt[((c_DM[a]*3+c_DM[b])*3+c_DM[c])*FDIM+f];
    p0[c] += w*(x0[a]*e0[b] + x1[a]*e1[b]);
    p1[c] += w*(x0[a]*e1[b] + x1[a]*e0[b]);
  }
  #pragma unroll
  for (int c=0;c<9;++c){
    ep[c*FDIM+f]     = p0[c];
    ep[(9+c)*FDIM+f] = p1[c];
    xp[c*FDIM+f]     = x0[c] + p0[c];
    xp[(9+c)*FDIM+f] = x1[c] + p1[c];
  }
}

// ---------------------------------------------------------------------------
// Xout = couple(U, V, Wp, maxdeg) [+ Add]
// ---------------------------------------------------------------------------
__global__ void k_tcouple(const float* __restrict__ U, const float* __restrict__ V,
                          const float* __restrict__ Wp,
                          const int* __restrict__ cgidx, const float* __restrict__ cgval,
                          const int* __restrict__ cgcnt,
                          float* __restrict__ Xout, const float* __restrict__ Add,
                          int maxdeg){
  __shared__ int   s_idx[768];
  __shared__ float s_val[768];
  int node = blockIdx.x, f = threadIdx.x;
  int nnz = *cgcnt;
  for (int t=f; t<nnz; t+=32){ s_idx[t]=cgidx[t]; s_val[t]=cgval[t]; }
  __syncthreads();
  const float* up = U + node*NROWS*FDIM;
  const float* vp = V + node*NROWS*FDIM;
  float u0[9],u1[9],v0[9],v1[9];
  #pragma unroll
  for (int a=0;a<9;++a){
    u0[a]=up[a*FDIM+f]; u1[a]=up[(9+a)*FDIM+f];
    v0[a]=vp[a*FDIM+f]; v1[a]=vp[(9+a)*FDIM+f];
  }
  float p0[9]={0,0,0,0,0,0,0,0,0}, p1[9]={0,0,0,0,0,0,0,0,0};
  for (int t=0;t<nnz;++t){
    int id=s_idx[t]; int a=(id>>8)&15, b=(id>>4)&15, c=id&15;
    float w = s_val[t]*Wp[((c_DM[a]*3+c_DM[b])*3+c_DM[c])*FDIM+f];
    p0[c] += w*(u0[a]*v0[b] + u1[a]*v1[b]);
    p1[c] += w*(u0[a]*v1[b] + u1[a]*v0[b]);
  }
  float* xo = Xout + node*NROWS*FDIM;
  #pragma unroll
  for (int c=0;c<9;++c){
    float m = (c_DM[c] <= maxdeg) ? 1.f : 0.f;
    float a0 = Add ? Add[node*NROWS*FDIM + c*FDIM+f]     : 0.f;
    float a1 = Add ? Add[node*NROWS*FDIM + (9+c)*FDIM+f] : 0.f;
    xo[c*FDIM+f]     = p0[c]*m + a0;
    xo[(9+c)*FDIM+f] = p1[c]*m + a1;
  }
}

// ---------------------------------------------------------------------------
// charges / atomic energies from scalar channel
// ---------------------------------------------------------------------------
__global__ void k_scal(const float* __restrict__ X, const float* __restrict__ Wq,
                       const float* __restrict__ We, const float* __restrict__ be,
                       const float* __restrict__ ebias, const int* __restrict__ Z,
                       float* __restrict__ Qc, float* __restrict__ AE){
  int n = blockIdx.x*blockDim.x + threadIdx.x;
  if (n >= BN) return;
  const float* xp = X + n*NROWS*FDIM;
  float q=0.f, e=0.f;
  #pragma unroll
  for (int f=0;f<FDIM;++f){ float s=xp[f]; q += s*Wq[f]; e += s*We[f]; }
  Qc[n]=q;
  AE[n]= e + be[0] + ebias[Z[n]];
}

// xd: keep scalar (p0,a0) and odd-parity l=1 (p1,a1..3), zero the rest
__global__ void k_xd(const float* __restrict__ X, float* __restrict__ XD){
  int node = blockIdx.x, f = threadIdx.x;
  const float* xp = X + node*NROWS*FDIM;
  float* dp = XD + node*NROWS*FDIM;
  #pragma unroll
  for (int r=0;r<NROWS;++r){
    float v = 0.f;
    if (r==0) v = xp[f];
    if (r>=10 && r<=12) v = xp[r*FDIM+f];
    dp[r*FDIM+f] = v;
  }
}

__global__ void k_adip(const float* __restrict__ XD, const float* __restrict__ wdip,
                       float* __restrict__ AD){
  int n = blockIdx.x*blockDim.x + threadIdx.x;
  if (n >= BN) return;
  #pragma unroll
  for (int c=0;c<3;++c){
    const float* p = XD + n*NROWS*FDIM + (10+c)*FDIM;
    float s=0.f;
    #pragma unroll
    for (int f=0;f<FDIM;++f) s += p[f]*wdip[f];
    AD[n*3+c]=s;
  }
}

// ---------------------------------------------------------------------------
// Per-batch reductions: COM, dipole, energy, Coulomb
// ---------------------------------------------------------------------------
__global__ void k_final(const float* __restrict__ pos, const float* __restrict__ Qc,
                        const float* __restrict__ AE, const float* __restrict__ AD,
                        const float* __restrict__ Er, float* __restrict__ out){
  __shared__ float buf[32];
  int b = blockIdx.x, t = threadIdx.x;
  int node = b*NNODE + t;
  float comv[3];
  for (int c=0;c<3;++c){
    buf[t]=pos[node*3+c]; __syncthreads();
    if (t==0){ float s=0; for (int k=0;k<32;++k) s+=buf[k]; buf[0]=s/32.f; }
    __syncthreads(); comv[c]=buf[0]; __syncthreads();
  }
  float q = Qc[node];
  float dip[3];
  for (int c=0;c<3;++c){
    buf[t] = q*(pos[node*3+c]-comv[c]) + AD[node*3+c]; __syncthreads();
    if (t==0){ float s=0; for (int k=0;k<32;++k) s+=buf[k]; buf[0]=s; }
    __syncthreads(); dip[c]=buf[0]; __syncthreads();
  }
  buf[t]=AE[node]; __syncthreads();
  if (t==0){ float s=0; for (int k=0;k<32;++k) s+=buf[k]; buf[0]=s; }
  __syncthreads();
  float esum = buf[0]; __syncthreads();
  float cs = 0.f;
  for (int e=t; e<EPB; e+=32){
    int i = e/(NNODE-1), tt = e%(NNODE-1);
    int j = (tt<i)? tt : tt+1;
    cs += Qc[b*NNODE+i]*Qc[b*NNODE+j]/(Er[b*EPB+e]+1e-10f);
  }
  buf[t]=cs; __syncthreads();
  if (t==0){
    float s=0; for (int k=0;k<32;++k) s+=buf[k];
    out[b] = esum + 0.5f*s*14.399645f;
    for (int c=0;c<3;++c) out[NBATCH + b*3 + c] = dip[c];
  }
}

// ---------------------------------------------------------------------------
extern "C" void kernel_launch(void* const* d_in, const int* in_sizes, int n_in,
                              void* d_out, int out_size, void* d_ws, size_t ws_size,
                              hipStream_t stream){
  (void)in_sizes; (void)n_in; (void)out_size; (void)ws_size;
  const int*   Z     = (const int*)  d_in[0];
  const float* pos   = (const float*)d_in[1];
  const float* Ef    = (const float*)d_in[2];
  const int*   dsti  = (const int*)  d_in[3];
  const int*   srci  = (const int*)  d_in[4];
  const float* embed = (const float*)d_in[6];
  const float* Wmp   = (const float*)d_in[7];
  const float* Wd    = (const float*)d_in[8];
  const float* bd    = (const float*)d_in[9];
  const float* Wt    = (const float*)d_in[10];
  const float* Wtd1  = (const float*)d_in[11];
  const float* Wtd2  = (const float*)d_in[12];
  const float* Wtdp  = (const float*)d_in[13];
  const float* Wh    = (const float*)d_in[14];
  const float* bh    = (const float*)d_in[15];
  const float* Wq    = (const float*)d_in[16];
  const float* Wdip1 = (const float*)d_in[17];
  const float* Wdip2 = (const float*)d_in[18];
  const float* Wdipp = (const float*)d_in[19];
  const float* wdip  = (const float*)d_in[20];
  const float* We    = (const float*)d_in[21];
  const float* be    = (const float*)d_in[22];
  const float* ebias = (const float*)d_in[23];
  float* out = (float*)d_out;

  // workspace carve-up (floats)
  float* w      = (float*)d_ws;
  float* CG     = w;            w += 768;
  int*   cgidx  = (int*)w;      w += 768;
  float* cgval  = w;            w += 768;
  int*   cgcnt  = (int*)w;      w += 16;
  float* Er     = w;            w += NEDGE;
  float* Esh    = w;            w += NEDGE*9;
  float* Erad   = w;            w += NEDGE*8;
  float* Xb     = w;            w += XSZ;
  float* Tb     = w;            w += XSZ;
  float* Yb     = w;            w += XSZ;
  float* XEF    = w;            w += XSZ;
  float* Ub     = w;            w += XSZ;
  float* Vb     = w;            w += XSZ;
  float* Qc     = w;            w += BN;
  float* AE     = w;            w += BN;
  float* AD     = w;            w += BN*3;

  k_cg    <<<1, 736, 0, stream>>>(CG);
  k_cgpack<<<1, 1,   0, stream>>>(CG, cgidx, cgval, cgcnt);
  k_edge  <<<(NEDGE+255)/256, 256, 0, stream>>>(pos, dsti, srci, Er, Esh, Erad);
  k_init  <<<BN, FDIM, 0, stream>>>(Z, embed, Ef, Xb, XEF);

  float* Xc = Xb; float* Tc = Tb;
  const int GB = MROWS/64;  // 288 blocks of 4 waves
  for (int it=0; it<3; ++it){
    int md = (it<2) ? 2 : 0;
    k_msg      <<<BN, FDIM, 0, stream>>>(Xc, Esh, Erad, Wmp + it*3*8*FDIM,
                                         cgidx, cgval, cgcnt, Yb, md);
    k_add_silu <<<BN, FDIM, 0, stream>>>(Xc, Yb);
    k_dense    <<<GB, 128, 0, stream>>>(Xc, Wd + it*FDIM*FDIM, bd + it*FDIM, Tc, 1);
    k_silu     <<<BN, FDIM, 0, stream>>>(Tc);
    { float* tmp=Xc; Xc=Tc; Tc=tmp; }
    k_couple_ef<<<BN, FDIM, 0, stream>>>(Xc, XEF, Wt + it*27*FDIM, cgidx, cgval, cgcnt);
    k_dense    <<<GB, 128, 0, stream>>>(Xc, Wtd1 + it*FDIM*FDIM, nullptr, Ub, 0);
    k_dense    <<<GB, 128, 0, stream>>>(Xc, Wtd2 + it*FDIM*FDIM, nullptr, Vb, 0);
    k_tcouple  <<<BN, FDIM, 0, stream>>>(Ub, Vb, Wtdp + it*27*FDIM,
                                         cgidx, cgval, cgcnt, Xc, Yb, 2);
  }
  for (int h=0; h<4; ++h){
    k_dense<<<GB, 128, 0, stream>>>(Xc, Wh + h*FDIM*FDIM, bh + h*FDIM, Tc, 1);
    k_silu <<<BN, FDIM, 0, stream>>>(Tc);
    { float* tmp=Xc; Xc=Tc; Tc=tmp; }
  }
  k_dense<<<GB, 128, 0, stream>>>(Xc, Wh + 4*FDIM*FDIM, bh + 4*FDIM, Tc, 1);
  { float* tmp=Xc; Xc=Tc; Tc=tmp; }

  k_scal   <<<(BN+255)/256, 256, 0, stream>>>(Xc, Wq, We, be, ebias, Z, Qc, AE);
  k_xd     <<<BN, FDIM, 0, stream>>>(Xc, Tc);
  k_dense  <<<GB, 128, 0, stream>>>(Tc, Wdip1, nullptr, Ub, 0);
  k_dense  <<<GB, 128, 0, stream>>>(Tc, Wdip2, nullptr, Vb, 0);
  k_tcouple<<<BN, FDIM, 0, stream>>>(Ub, Vb, Wdipp, cgidx, cgval, cgcnt, Yb, nullptr, 1);
  k_adip   <<<(BN+255)/256, 256, 0, stream>>>(Yb, wdip, AD);
  k_final  <<<NBATCH, 32, 0, stream>>>(pos, Qc, AE, AD, Er, out);
}
